// Model_43439299231970
// MI455X (gfx1250) — matrix operand order repaired and checked by reading, not compiled, not run on hardware
//
#include <hip/hip_runtime.h>
#include <cstddef>
#include <cstdint>

// ---------------- model dimensions (match reference) ----------------
#define B_DIM   64
#define GS_DIM  512
#define NB      30000
#define EE      480000
#define EA      (EE + NB)      // edges + self loops
#define FNODE   64
#define PROJ    128
#define MLP     256
#define HEADS   4
#define HID     64
#define HIDC    256            // HEADS*HID
#define EEMB    32
#define OUT_EMB 256
#define TRUNK   512
#define POLICY  4096
#define VALUE_D 4
#define EPSV    1e-5f
#define NEG_SLOPE 0.2f

typedef __bf16 bf16t;
typedef __attribute__((ext_vector_type(2)))  __bf16   bf16x2;
typedef __attribute__((ext_vector_type(16))) __bf16   v16bf;
typedef __attribute__((ext_vector_type(8)))  float    v8f;
typedef __attribute__((ext_vector_type(8)))  unsigned v8u;

__device__ __forceinline__ unsigned pack_bf16(float a, float b)
{
    bf16x2 p;
    p.x = (bf16t)a;
    p.y = (bf16t)b;              // lowers to v_cvt_pk_bf16_f32
    return __builtin_bit_cast(unsigned, p);
}

// =====================================================================
// Generic GEMM: Y[M,N] = X[M,K] @ W[K,N] + bias   (bf16 WMMA, f32 accum)
// Requires K % 32 == 0. M,N arbitrary (bounds checked).
// Block: 128 threads = 4 waves. Wave w -> 16-row strip, 64 cols:
//   4 accumulators, 4 WMMAs per K-step sharing one A fragment.
// LDS staged as packed bf16x2 dwords -> fragment build = 8x ds_load_b32.
// =====================================================================
__global__ void gemm_bf16_kernel(const float* __restrict__ X,
                                 const float* __restrict__ W,
                                 const float* __restrict__ bias,
                                 float* __restrict__ Y,
                                 int M, int K, int N)
{
    // A: 64 rows x 16 K-pairs ; B: 16 K-pairs x 64 cols (padded)
    __shared__ unsigned sA[64][17];
    __shared__ unsigned sB[16][68];

    const int tid    = threadIdx.x;    // 0..127
    const int wave   = tid >> 5;       // 0..3
    const int lane   = tid & 31;
    const int laneHi = lane >> 4;      // 0|1
    const int lane16 = lane & 15;
    const int rowBase = blockIdx.x * 64;
    const int colBase = blockIdx.y * 64;

    v8f acc0 = {}, acc1 = {}, acc2 = {}, acc3 = {};

    for (int k0 = 0; k0 < K; k0 += 32) {
        // ---- stage A tile (64 rows x 16 pairs), float2 loads ----
        #pragma unroll
        for (int i = 0; i < 8; ++i) {
            int e = tid + i * 128;                 // 0..1023
            int r = e >> 4, p = e & 15;
            int gr = rowBase + r;
            float2 v = make_float2(0.f, 0.f);
            if (gr < M) {
                v = *(const float2*)(X + (size_t)gr * K + k0 + 2 * p);
                if (k0 + 32 < K)                   // prefetch next K-step
                    __builtin_prefetch(X + (size_t)gr * K + k0 + 32 + 2 * p, 0, 3);
            }
            sA[r][p] = pack_bf16(v.x, v.y);
        }
        // ---- stage B tile (16 pairs x 64 cols) ----
        #pragma unroll
        for (int i = 0; i < 8; ++i) {
            int e = tid + i * 128;                 // 0..1023
            int p = e >> 6, c = e & 63;
            int gc = colBase + c;
            float a = 0.f, b = 0.f;
            if (gc < N) {
                a = W[(size_t)(k0 + 2 * p) * N + gc];
                b = W[(size_t)(k0 + 2 * p + 1) * N + gc];
            }
            sB[p][c] = pack_bf16(a, b);
        }
        __syncthreads();

        // ---- A fragment: row m, pair pi = vv + laneHi*8 ----
        const int m = wave * 16 + lane16;
        v8u au;
        #pragma unroll
        for (int vv = 0; vv < 8; ++vv) {
            int pi = vv + (laneHi ? 8 : 0);
            au[vv] = sA[m][pi];
        }
        v16bf afrag = __builtin_bit_cast(v16bf, au);

        // ---- 4 B fragments / 4 WMMAs reusing afrag ----
        v8u bu;
        #pragma unroll
        for (int vv = 0; vv < 8; ++vv) bu[vv] = sB[vv + (laneHi ? 8 : 0)][0 * 16 + lane16];
        acc0 = __builtin_amdgcn_wmma_f32_16x16x32_bf16(false, afrag, false,
                   __builtin_bit_cast(v16bf, bu), (short)0, acc0, false, false);
        #pragma unroll
        for (int vv = 0; vv < 8; ++vv) bu[vv] = sB[vv + (laneHi ? 8 : 0)][1 * 16 + lane16];
        acc1 = __builtin_amdgcn_wmma_f32_16x16x32_bf16(false, afrag, false,
                   __builtin_bit_cast(v16bf, bu), (short)0, acc1, false, false);
        #pragma unroll
        for (int vv = 0; vv < 8; ++vv) bu[vv] = sB[vv + (laneHi ? 8 : 0)][2 * 16 + lane16];
        acc2 = __builtin_amdgcn_wmma_f32_16x16x32_bf16(false, afrag, false,
                   __builtin_bit_cast(v16bf, bu), (short)0, acc2, false, false);
        #pragma unroll
        for (int vv = 0; vv < 8; ++vv) bu[vv] = sB[vv + (laneHi ? 8 : 0)][3 * 16 + lane16];
        acc3 = __builtin_amdgcn_wmma_f32_16x16x32_bf16(false, afrag, false,
                   __builtin_bit_cast(v16bf, bu), (short)0, acc3, false, false);

        __syncthreads();
    }

    // ---- store: lanes 0-15 -> M=g, lanes 16-31 -> M=g+8 ; N = lane16 ----
    const int mBase = rowBase + wave * 16 + (laneHi ? 8 : 0);
    v8f accs[4] = { acc0, acc1, acc2, acc3 };
    #pragma unroll
    for (int t = 0; t < 4; ++t) {
        int ncol = colBase + t * 16 + lane16;
        if (ncol < N) {
            float bv = bias ? bias[ncol] : 0.0f;
            #pragma unroll
            for (int g = 0; g < 8; ++g) {
                int mrow = mBase + g;
                if (mrow < M) Y[(size_t)mrow * N + ncol] = accs[t][g] + bv;
            }
        }
    }
}

// =====================================================================
// BatchNorm (batch statistics) : one block per column
// =====================================================================
__global__ void bn_stats_kernel(const float* __restrict__ X,
                                float* __restrict__ mean,
                                float* __restrict__ rstd, int M, int C)
{
    int c = blockIdx.x, tid = threadIdx.x;
    float s = 0.f, ss = 0.f;
    for (int r = tid; r < M; r += 256) {
        float v = X[(size_t)r * C + c];
        s += v; ss += v * v;
    }
    __shared__ float sh[256], sh2[256];
    sh[tid] = s; sh2[tid] = ss; __syncthreads();
    for (int o = 128; o > 0; o >>= 1) {
        if (tid < o) { sh[tid] += sh[tid + o]; sh2[tid] += sh2[tid + o]; }
        __syncthreads();
    }
    if (tid == 0) {
        float m = sh[0] / (float)M;
        float v = sh2[0] / (float)M - m * m;
        mean[c] = m;
        rstd[c] = rsqrtf(v + EPSV);
    }
}

__global__ void bn_apply_kernel(float* __restrict__ X,
                                const float* __restrict__ mean,
                                const float* __restrict__ rstd,
                                const float* __restrict__ g,
                                const float* __restrict__ b,
                                long long total, int cmask, int relu)
{
    long long idx = (long long)blockIdx.x * blockDim.x + threadIdx.x;
    if (idx >= total) return;
    int c = (int)(idx & cmask);                 // C is a power of two
    float y = (X[idx] - mean[c]) * rstd[c] * g[c] + b[c];
    if (relu) y = fmaxf(y, 0.0f);
    X[idx] = y;
}

// =====================================================================
// misc elementwise
// =====================================================================
__global__ void zero_kernel(float* p, long long n)
{
    long long i = (long long)blockIdx.x * blockDim.x + threadIdx.x;
    if (i < n) p[i] = 0.0f;
}

__global__ void add_relu_kernel(float* a, const float* c, long long n)
{
    long long i = (long long)blockIdx.x * blockDim.x + threadIdx.x;
    if (i < n) a[i] = fmaxf(a[i] + c[i], 0.0f);
}

__global__ void tanh_kernel(float* p, long long n)
{
    long long i = (long long)blockIdx.x * blockDim.x + threadIdx.x;
    if (i < n) p[i] = tanhf(p[i]);
}

__global__ void concat_kernel(const float* gs, const float* mp, float* fused)
{
    int idx = blockIdx.x * blockDim.x + threadIdx.x;
    if (idx >= B_DIM * TRUNK) return;
    int b = idx / TRUNK, j = idx % TRUNK;
    fused[idx] = (j < MLP) ? gs[b * MLP + j] : mp[b * OUT_EMB + (j - MLP)];
}

// =====================================================================
// edge preparation: embedding gather + self-loop mean attrs + adjacency
// =====================================================================
__global__ void edge_gather_kernel(const float* __restrict__ emb,
                                   const int* __restrict__ cat,
                                   const int* __restrict__ ei,
                                   float* __restrict__ e_all,
                                   float* __restrict__ cnt)
{
    long long idx = (long long)blockIdx.x * blockDim.x + threadIdx.x;
    if (idx >= (long long)EE * EEMB) return;
    int e = (int)(idx >> 5), k = (int)(idx & 31);
    float v = emb[cat[e] * EEMB + k];
    e_all[idx] = v;
    int d = ei[EE + e];
    atomicAdd(&e_all[((size_t)(EE + d)) * EEMB + k], v);
    if (k == 0) atomicAdd(&cnt[d], 1.0f);
}

__global__ void loop_div_kernel(float* __restrict__ e_all,
                                const float* __restrict__ cnt)
{
    long long idx = (long long)blockIdx.x * blockDim.x + threadIdx.x;
    if (idx >= (long long)NB * EEMB) return;
    int i = (int)(idx >> 5);
    float c = cnt[i]; c = (c > 1.0f) ? c : 1.0f;
    e_all[(size_t)EE * EEMB + idx] /= c;
}

__global__ void build_adj_kernel(const int* __restrict__ ei,
                                 int* __restrict__ srcA, int* __restrict__ dstA)
{
    int idx = blockIdx.x * blockDim.x + threadIdx.x;
    if (idx >= EA) return;
    if (idx < EE) { srcA[idx] = ei[idx]; dstA[idx] = ei[EE + idx]; }
    else          { srcA[idx] = idx - EE; dstA[idx] = idx - EE; }
}

// =====================================================================
// GATv2 attention (e.We fused into the logit kernel to avoid a 0.5 GB buf)
// =====================================================================
__device__ __forceinline__ unsigned ord_encode(float f)
{
    unsigned u = __float_as_uint(f);
    return (u & 0x80000000u) ? ~u : (u | 0x80000000u);
}
__device__ __forceinline__ float ord_decode(unsigned u)
{
    unsigned bits = (u & 0x80000000u) ? (u ^ 0x80000000u) : ~u;
    return __uint_as_float(bits);
}

__global__ void gat_logits_kernel(const float* __restrict__ xl,
                                  const float* __restrict__ xr,
                                  const float* __restrict__ e_all,
                                  const float* __restrict__ we,
                                  const float* __restrict__ att,
                                  const int* __restrict__ srcA,
                                  const int* __restrict__ dstA,
                                  float* __restrict__ logits,
                                  unsigned* __restrict__ nmax)
{
    long long idx = (long long)blockIdx.x * blockDim.x + threadIdx.x;
    if (idx >= (long long)EA * HEADS) return;
    int h = (int)(idx & 3);
    long long e = idx >> 2;
    int s = srcA[e], d = dstA[e];

    float ebuf[EEMB];
    const float* ea = e_all + (size_t)e * EEMB;
    #pragma unroll
    for (int k = 0; k < EEMB; ++k) ebuf[k] = ea[k];

    const float* xls  = xl  + (size_t)s * HIDC + h * HID;
    const float* xrd  = xr  + (size_t)d * HIDC + h * HID;
    const float* attn = att + h * HID;

    float logit = 0.0f;
    for (int c = 0; c < HID; ++c) {
        float et = 0.0f;
        const float* wcol = we + h * HID + c;
        #pragma unroll
        for (int k = 0; k < EEMB; ++k) et += ebuf[k] * wcol[k * HIDC];
        float m = xls[c] + xrd[c] + et;
        m = (m > 0.0f) ? m : NEG_SLOPE * m;       // leaky_relu 0.2
        logit += m * attn[c];
    }
    logits[idx] = logit;
    atomicMax(&nmax[(size_t)d * HEADS + h], ord_encode(logit));
}

__global__ void gat_exp_kernel(const float* __restrict__ logits,
                               const unsigned* __restrict__ nmax,
                               const int* __restrict__ dstA,
                               float* __restrict__ aexp,
                               float* __restrict__ den)
{
    long long idx = (long long)blockIdx.x * blockDim.x + threadIdx.x;
    if (idx >= (long long)EA * HEADS) return;
    int h = (int)(idx & 3);
    long long e = idx >> 2;
    int d = dstA[e];
    float mx = ord_decode(nmax[(size_t)d * HEADS + h]);
    float a = __expf(logits[idx] - mx);
    aexp[idx] = a;
    atomicAdd(&den[(size_t)d * HEADS + h], a);
}

__global__ void gat_aggr_kernel(const float* __restrict__ xl,
                                const float* __restrict__ aexp,
                                const float* __restrict__ den,
                                const int* __restrict__ srcA,
                                const int* __restrict__ dstA,
                                float* __restrict__ out)
{
    long long idx = (long long)blockIdx.x * blockDim.x + threadIdx.x;
    if (idx >= (long long)EA * HIDC) return;
    int c = (int)(idx & 255);
    long long e = idx >> 8;
    int h = c >> 6;
    int s = srcA[e], d = dstA[e];
    float alpha = aexp[e * HEADS + h] / den[(size_t)d * HEADS + h];
    atomicAdd(&out[(size_t)d * HIDC + c], xl[(size_t)s * HIDC + c] * alpha);
}

// =====================================================================
// global mean pool
// =====================================================================
__global__ void pool_acc_kernel(const float* __restrict__ h,
                                const int* __restrict__ nbidx,
                                float* __restrict__ mapsum,
                                float* __restrict__ gcnt)
{
    long long idx = (long long)blockIdx.x * blockDim.x + threadIdx.x;
    if (idx >= (long long)NB * OUT_EMB) return;
    int c = (int)(idx & 255);
    int i = (int)(idx >> 8);
    int b = nbidx[i];
    atomicAdd(&mapsum[b * OUT_EMB + c], h[idx]);
    if (c == 0) atomicAdd(&gcnt[b], 1.0f);
}

__global__ void pool_div_kernel(float* __restrict__ mapsum,
                                const float* __restrict__ gcnt)
{
    int idx = blockIdx.x * blockDim.x + threadIdx.x;
    if (idx >= B_DIM * OUT_EMB) return;
    int b = idx >> 8;
    float c = gcnt[b]; c = (c > 1.0f) ? c : 1.0f;
    mapsum[idx] /= c;
}

// =====================================================================
// row-wise log_softmax (block per row)
// =====================================================================
__global__ void log_softmax_kernel(float* __restrict__ X, int C)
{
    int b = blockIdx.x, tid = threadIdx.x;
    float* row = X + (size_t)b * C;
    __shared__ float sh[256];

    float mx = -3.4e38f;
    for (int c = tid; c < C; c += 256) mx = fmaxf(mx, row[c]);
    sh[tid] = mx; __syncthreads();
    for (int o = 128; o > 0; o >>= 1) {
        if (tid < o) sh[tid] = fmaxf(sh[tid], sh[tid + o]);
        __syncthreads();
    }
    float mval = sh[0]; __syncthreads();

    float s = 0.0f;
    for (int c = tid; c < C; c += 256) s += __expf(row[c] - mval);
    sh[tid] = s; __syncthreads();
    for (int o = 128; o > 0; o >>= 1) {
        if (tid < o) sh[tid] += sh[tid + o];
        __syncthreads();
    }
    float lse = mval + logf(sh[0]);
    __syncthreads();
    for (int c = tid; c < C; c += 256) row[c] -= lse;
}

// =====================================================================
// host side
// =====================================================================
static inline unsigned nblk(long long n, int t) { return (unsigned)((n + t - 1) / t); }

static void launch_gemm(hipStream_t s, const float* X, const float* W,
                        const float* bias, float* Y, int M, int K, int N)
{
    dim3 grid((M + 63) / 64, (N + 63) / 64);
    gemm_bf16_kernel<<<grid, dim3(128), 0, s>>>(X, W, bias, Y, M, K, N);
}

static void launch_bn(hipStream_t s, float* X, const float* g, const float* b,
                      float* mean, float* rstd, int M, int C, int relu)
{
    bn_stats_kernel<<<C, 256, 0, s>>>(X, mean, rstd, M, C);
    long long tot = (long long)M * C;
    bn_apply_kernel<<<nblk(tot, 256), 256, 0, s>>>(X, mean, rstd, g, b, tot, C - 1, relu);
}

static void launch_zero(hipStream_t s, float* p, long long n)
{
    zero_kernel<<<nblk(n, 256), 256, 0, s>>>(p, n);
}

extern "C" void kernel_launch(void* const* d_in, const int* in_sizes, int n_in,
                              void* d_out, int out_size, void* d_ws, size_t ws_size,
                              hipStream_t stream)
{
    (void)in_sizes; (void)n_in; (void)out_size; (void)ws_size;
    const float* x_gs  = (const float*)d_in[0];
    const float* x_map = (const float*)d_in[1];
    const int*   ei    = (const int*)d_in[2];   // [2,E]
    const int*   nbidx = (const int*)d_in[3];   // [N]
    const int*   ecat  = (const int*)d_in[4];   // [E]
    auto F = [&](int i) { return (const float*)d_in[i]; };

    // ---- params (flattened in setup_inputs insertion order) ----
    const float* gs1_w = F(5),  *gs1_b = F(6),  *bn_gs1_g = F(7),  *bn_gs1_b = F(8);
    const float* gs2_w = F(9),  *gs2_b = F(10), *bn_gs2_g = F(11), *bn_gs2_b = F(12);
    const float* np_w  = F(13), *np_b  = F(14), *bn_nd_g  = F(15), *bn_nd_b  = F(16);
    const float* edge_emb = F(17);
    // gat[i]: wl,bl,wr,br,we,att,bias  at 18+7*i ; gat_bn[i]: g,b at 39+2*i
    const float* fp_w = F(45), *fp_b = F(46), *bn_f_g = F(47), *bn_f_b = F(48);
    const float* fu_w = F(49), *fu_b = F(50), *bn_u_g = F(51), *bn_u_b = F(52);
    // res[r]: fc1.w,fc1.b,bn1.g,bn1.b,fc2.w,fc2.b,bn2.g,bn2.b at 53+8*r
    const float* po_w = F(93), *po_b = F(94), *va_w = F(95), *va_b = F(96);

    // ---- workspace carve-up ----
    char* ws = (char*)d_ws;
    size_t off = 0;
    auto alloc = [&](size_t bytes) -> char* {
        char* p = ws + off;
        off = (off + bytes + 255) & ~(size_t)255;
        return p;
    };
    float*    nodeA  = (float*)alloc((size_t)NB * HIDC * 4);   // h / gat out
    float*    nodeB  = (float*)alloc((size_t)NB * HIDC * 4);   // xl
    float*    nodeC  = (float*)alloc((size_t)NB * HIDC * 4);   // xr
    float*    e_all  = (float*)alloc((size_t)EA * EEMB * 4);
    float*    logits = (float*)alloc((size_t)EA * HEADS * 4);
    float*    aexp   = (float*)alloc((size_t)EA * HEADS * 4);
    unsigned* nmax   = (unsigned*)alloc((size_t)NB * HEADS * 4);
    float*    den    = (float*)alloc((size_t)NB * HEADS * 4);
    float*    cnt    = (float*)alloc((size_t)NB * 4);
    int*      srcA   = (int*)alloc((size_t)EA * 4);
    int*      dstA   = (int*)alloc((size_t)EA * 4);
    float*    mean   = (float*)alloc(4096 * 4);
    float*    rstd   = (float*)alloc(4096 * 4);
    float*    gsA    = (float*)alloc((size_t)B_DIM * MLP * 4);
    float*    gsB    = (float*)alloc((size_t)B_DIM * MLP * 4);
    float*    mapsum = (float*)alloc((size_t)B_DIM * OUT_EMB * 4);
    float*    gcnt   = (float*)alloc((size_t)B_DIM * 4);
    float*    fused  = (float*)alloc((size_t)B_DIM * TRUNK * 4);
    float*    trunkA = (float*)alloc((size_t)B_DIM * TRUNK * 4);
    float*    trunkB = (float*)alloc((size_t)B_DIM * TRUNK * 4);
    float*    trunkC = (float*)alloc((size_t)B_DIM * TRUNK * 4);

    // ---- game-state MLP branch (linear biases cancel in BN, kept anyway) ----
    launch_gemm(stream, x_gs, gs1_w, gs1_b, gsA, B_DIM, GS_DIM, MLP);
    launch_bn(stream, gsA, bn_gs1_g, bn_gs1_b, mean, rstd, B_DIM, MLP, 1);
    launch_gemm(stream, gsA, gs2_w, gs2_b, gsB, B_DIM, MLP, MLP);
    launch_bn(stream, gsB, bn_gs2_g, bn_gs2_b, mean, rstd, B_DIM, MLP, 1);

    // ---- node init projection ----
    launch_gemm(stream, x_map, np_w, np_b, nodeA, NB, FNODE, PROJ);
    launch_bn(stream, nodeA, bn_nd_g, bn_nd_b, mean, rstd, NB, PROJ, 1);

    // ---- edge embedding + self-loop mean attrs + adjacency ----
    launch_zero(stream, cnt, NB);
    launch_zero(stream, e_all + (size_t)EE * EEMB, (long long)NB * EEMB);
    edge_gather_kernel<<<nblk((long long)EE * EEMB, 256), 256, 0, stream>>>(
        edge_emb, ecat, ei, e_all, cnt);
    loop_div_kernel<<<nblk((long long)NB * EEMB, 256), 256, 0, stream>>>(e_all, cnt);
    build_adj_kernel<<<nblk(EA, 256), 256, 0, stream>>>(ei, srcA, dstA);

    // ---- GATv2 stack ----
    int din = PROJ;
    for (int layer = 0; layer < 3; ++layer) {
        const float* wl  = F(18 + 7 * layer + 0);
        const float* bl  = F(18 + 7 * layer + 1);
        const float* wr  = F(18 + 7 * layer + 2);
        const float* br  = F(18 + 7 * layer + 3);
        const float* we  = F(18 + 7 * layer + 4);
        const float* att = F(18 + 7 * layer + 5);
        // gat bias (index +6) is a per-column constant into BN -> cancels
        const float* bng = F(39 + 2 * layer + 0);
        const float* bnb = F(39 + 2 * layer + 1);

        launch_gemm(stream, nodeA, wl, bl, nodeB, NB, din, HIDC);   // xl
        launch_gemm(stream, nodeA, wr, br, nodeC, NB, din, HIDC);   // xr

        launch_zero(stream, nodeA, (long long)NB * HIDC);           // out accum
        launch_zero(stream, (float*)nmax, (long long)NB * HEADS);   // ord(-inf) id
        launch_zero(stream, den, (long long)NB * HEADS);

        gat_logits_kernel<<<nblk((long long)EA * HEADS, 256), 256, 0, stream>>>(
            nodeB, nodeC, e_all, we, att, srcA, dstA, logits, nmax);
        gat_exp_kernel<<<nblk((long long)EA * HEADS, 256), 256, 0, stream>>>(
            logits, nmax, dstA, aexp, den);
        gat_aggr_kernel<<<nblk((long long)EA * HIDC, 256), 256, 0, stream>>>(
            nodeB, aexp, den, srcA, dstA, nodeA);

        launch_bn(stream, nodeA, bng, bnb, mean, rstd, NB, HIDC, 1);
        din = HIDC;
    }

    // ---- final projection + BN + ReLU ----
    launch_gemm(stream, nodeA, fp_w, fp_b, nodeB, NB, HIDC, OUT_EMB);
    launch_bn(stream, nodeB, bn_f_g, bn_f_b, mean, rstd, NB, OUT_EMB, 1);

    // ---- global mean pool ----
    launch_zero(stream, mapsum, (long long)B_DIM * OUT_EMB);
    launch_zero(stream, gcnt, B_DIM);
    pool_acc_kernel<<<nblk((long long)NB * OUT_EMB, 256), 256, 0, stream>>>(
        nodeB, nbidx, mapsum, gcnt);
    pool_div_kernel<<<nblk(B_DIM * OUT_EMB, 256), 256, 0, stream>>>(mapsum, gcnt);

    // ---- fusion + residual trunk ----
    concat_kernel<<<nblk(B_DIM * TRUNK, 256), 256, 0, stream>>>(gsB, mapsum, fused);
    launch_gemm(stream, fused, fu_w, fu_b, trunkA, B_DIM, TRUNK, TRUNK);
    launch_bn(stream, trunkA, bn_u_g, bn_u_b, mean, rstd, B_DIM, TRUNK, 1);

    for (int r = 0; r < 5; ++r) {
        const float* f1w = F(53 + 8 * r + 0), *f1b = F(53 + 8 * r + 1);
        const float* b1g = F(53 + 8 * r + 2), *b1b = F(53 + 8 * r + 3);
        const float* f2w = F(53 + 8 * r + 4), *f2b = F(53 + 8 * r + 5);
        const float* b2g = F(53 + 8 * r + 6), *b2b = F(53 + 8 * r + 7);
        launch_gemm(stream, trunkA, f1w, f1b, trunkB, B_DIM, TRUNK, TRUNK);
        launch_bn(stream, trunkB, b1g, b1b, mean, rstd, B_DIM, TRUNK, 1);
        launch_gemm(stream, trunkB, f2w, f2b, trunkC, B_DIM, TRUNK, TRUNK);
        launch_bn(stream, trunkC, b2g, b2b, mean, rstd, B_DIM, TRUNK, 0);
        add_relu_kernel<<<nblk((long long)B_DIM * TRUNK, 256), 256, 0, stream>>>(
            trunkA, trunkC, (long long)B_DIM * TRUNK);
    }

    // ---- heads (write directly into d_out: [64*4096 policy | 64*4 value]) ----
    float* pol = (float*)d_out;
    float* val = pol + (size_t)B_DIM * POLICY;
    launch_gemm(stream, trunkA, po_w, po_b, pol, B_DIM, TRUNK, POLICY);
    log_softmax_kernel<<<B_DIM, 256, 0, stream>>>(pol, POLICY);
    launch_gemm(stream, trunkA, va_w, va_b, val, B_DIM, TRUNK, VALUE_D);
    tanh_kernel<<<1, 256, 0, stream>>>(val, (long long)B_DIM * VALUE_D);
}
